// TreeLSTMEncoder_22247930593472
// MI455X (gfx1250) — matrix operand order repaired and checked by reading, not compiled, not run on hardware
//
#include <hip/hip_runtime.h>

// ---------------------------------------------------------------------------
// TreeLSTM encoder for MI455X (gfx1250), wave32 + WMMA bf16 + async LDS copy.
//
// Per tree level (M = 128*2^lvl rows):
//   gates = x @ [Wi|Wo|Wu|Wf] + h0 @ [Ui|Uo|Uu|Uf] + h1 @ [Ui|Uo|Uu|Uf]
// i/o/u contributions chain into one WMMA accumulator; forget gate keeps
// x/h0/h1 parts split (per-child forget). All matmuls use
// v_wmma_f32_16x16x32_bf16. Embedding table + weights are converted to bf16
// once; h is stored bf16 (only a WMMA A-operand), cell state c stays fp32.
// B-tiles (8 gates x 16 cols x K) are staged into LDS with
// global_load_async_to_lds_b128 (ASYNCcnt) and shared by the 4 waves of a
// block; each wave computes two 16x16 M-subtiles (24 WMMAs per K=32 step).
// ---------------------------------------------------------------------------

typedef __bf16 bf16_t;
typedef __attribute__((ext_vector_type(16))) __bf16 v16bf;
typedef __attribute__((ext_vector_type(8)))  __bf16 v8bf;
typedef __attribute__((ext_vector_type(8)))  float  v8f;
typedef __attribute__((ext_vector_type(4)))  float  v4f;

#define BQ   128   // batch
#define EQ   512   // embedding dim
#define HQ   512   // hidden dim
#define NSEQ 511   // nodes per tree
#define MATQ 262144  // 512*512 elements per packed weight matrix

__device__ __forceinline__ v8f wmma_bf16(v16bf a, v16bf b, v8f c) {
  return __builtin_amdgcn_wmma_f32_16x16x32_bf16(false, a, false, b,
                                                 (short)0, c, false, false);
}

// A-matrix 16x32 bf16 fragment (ISA 7.12.2): lanes 0-15 hold K=[k0..k0+7] and
// [k0+16..k0+23]; lanes 16-31 hold K=[k0+8..k0+15] and [k0+24..k0+31].
__device__ __forceinline__ v16bf fragA_bf16(const bf16_t* rowp, int k0, int laneHi) {
  const bf16_t* p = rowp + k0 + (laneHi ? 8 : 0);
  v8bf lo = *(const v8bf*)p;
  v8bf hi = *(const v8bf*)(p + 16);
  return __builtin_shufflevector(lo, hi, 0,1,2,3,4,5,6,7,8,9,10,11,12,13,14,15);
}

// B-matrix 32x16 bf16 fragment out of the LDS-staged [gate][n][K] tile:
// lane holds column N; lanes 0-15 read K=[k0..k0+7] and [k0+16..k0+23], lanes 16-31
// K=[k0+8..k0+15] and [k0+24..k0+31] -> 2 x ds_load_b128.
__device__ __forceinline__ v16bf fragB_lds(const bf16_t* tile, int gate, int laneN,
                                           int ldK, int k0, int laneHi) {
  const bf16_t* p = tile + (size_t)(gate * 16 + laneN) * ldK + k0 + (laneHi ? 8 : 0);
  v8bf lo = *(const v8bf*)p;
  v8bf hi = *(const v8bf*)(p + 16);
  return __builtin_shufflevector(lo, hi, 0,1,2,3,4,5,6,7,8,9,10,11,12,13,14,15);
}

// gfx1250 async global->LDS copy, 16 bytes per lane; tracked by ASYNCcnt.
__device__ __forceinline__ void async_copy_b128(bf16_t* lds_dst, const bf16_t* gsrc) {
  unsigned int lds_off = (unsigned int)(unsigned long long)lds_dst;
  asm volatile("global_load_async_to_lds_b128 %0, %1, off"
               :: "v"(lds_off), "v"(gsrc) : "memory");
}
__device__ __forceinline__ void wait_async0() {
#if __has_builtin(__builtin_amdgcn_s_wait_asynccnt)
  __builtin_amdgcn_s_wait_asynccnt(0);
#else
  asm volatile("s_wait_asynccnt 0x0" ::: "memory");
#endif
}

__device__ __forceinline__ float fsig(float x) {
  return 1.0f / (1.0f + __expf(-x));
}

// ---------------------------------------------------------------------------
// Pack weights: out[mat][n][k] = W[k*512 + n] as bf16.
// mat 0..3 = Wi,Wo,Wu,Wf -> WxT ; 4..7 = Ui,Uo,Uu,Uf -> UT (contiguous with
// WxT, forming "Wall"); 8 = Wp -> WpT.
// ---------------------------------------------------------------------------
__global__ void __launch_bounds__(256) cvt_weights_kernel(
    const float* __restrict__ Wi, const float* __restrict__ Wo,
    const float* __restrict__ Wu, const float* __restrict__ Wf,
    const float* __restrict__ Ui, const float* __restrict__ Uo,
    const float* __restrict__ Uu, const float* __restrict__ Uf,
    const float* __restrict__ Wp,
    bf16_t* __restrict__ Wall, bf16_t* __restrict__ WpT)
{
  const int t   = blockIdx.x * blockDim.x + threadIdx.x;
  const int mat = t >> 18;
  const int idx = t & (MATQ - 1);
  const int n   = idx >> 9;
  const int k   = idx & 511;
  const int s   = k * 512 + n;
  float v; bf16_t* dst;
  switch (mat) {
    case 0: v = Wi[s]; dst = Wall + 0 * MATQ; break;
    case 1: v = Wo[s]; dst = Wall + 1 * MATQ; break;
    case 2: v = Wu[s]; dst = Wall + 2 * MATQ; break;
    case 3: v = Wf[s]; dst = Wall + 3 * MATQ; break;
    case 4: v = Ui[s]; dst = Wall + 4 * MATQ; break;
    case 5: v = Uo[s]; dst = Wall + 5 * MATQ; break;
    case 6: v = Uu[s]; dst = Wall + 6 * MATQ; break;
    case 7: v = Uf[s]; dst = Wall + 7 * MATQ; break;
    default: v = Wp[s]; dst = WpT;            break;
  }
  dst[idx] = (bf16_t)v;
}

// Convert the whole embedding table fp32 -> bf16 (one streaming pass).
__global__ void __launch_bounds__(256) cvt_emb_kernel(
    const float* __restrict__ emb, bf16_t* __restrict__ embB)
{
  const size_t base = ((size_t)blockIdx.x * blockDim.x + threadIdx.x) * 8;
  v4f a = *(const v4f*)(emb + base);
  v4f b = *(const v4f*)(emb + base + 4);
  v8bf r;
#pragma unroll
  for (int i = 0; i < 4; ++i) { r[i] = (bf16_t)a[i]; r[4 + i] = (bf16_t)b[i]; }
  *(v8bf*)(embB + base) = r;
}

// ---------------------------------------------------------------------------
// Leaf level (lvl=8): c = sig(x@Wi+bi)*tanh(x@Wu+bu), h = sig(x@Wo+bo)*tanh(c)
// Block = 4 waves; wave owns two 16x16 M-subtiles; B (3 gates) staged in LDS.
// ---------------------------------------------------------------------------
__global__ void __launch_bounds__(128) leaf_kernel(
    const int* __restrict__ word_ids, const bf16_t* __restrict__ embB,
    const bf16_t* __restrict__ Wall,
    const float* __restrict__ bi, const float* __restrict__ bo,
    const float* __restrict__ bu,
    bf16_t* __restrict__ hOut, float* __restrict__ cOut)
{
  extern __shared__ char smem[];
  bf16_t* tile = (bf16_t*)smem;                 // [3][16][512] = 48 KB

  const int lane   = threadIdx.x & 31;
  const int wid    = threadIdx.x >> 5;
  const int laneN  = lane & 15;
  const int laneHi = lane >> 4;
  const int mTile  = blockIdx.x * 4 + wid;      // 32 rows per wave
  const int nTile  = blockIdx.y;
  const int lg = 8, nNodes = 256, start = 255;
  const int nG = nTile * 16 + laneN;

  // Async-stage B tile: gates {Wi,Wo,Wu}, 16 cols, full K=512.
  for (int idx = threadIdx.x; idx < 3 * 16 * 64; idx += 128) {
    const int g   = idx >> 10;
    const int rem = idx & 1023;
    const int row = rem >> 6;
    const int ck  = rem & 63;
    async_copy_b128(tile + ((size_t)(g * 16 + row) * 512 + ck * 8),
                    Wall + (size_t)g * MATQ + (size_t)(nTile * 16 + row) * 512 + ck * 8);
  }

  // Per-subtile embedding rows.
  const bf16_t* xrow[2];
#pragma unroll
  for (int r = 0; r < 2; ++r) {
    const int rowA  = mTile * 32 + r * 16 + laneN;
    const int bA    = rowA >> lg;
    const int nodeA = rowA & (nNodes - 1);
    xrow[r] = embB + (size_t)word_ids[bA * NSEQ + start + nodeA] * EQ;
    __builtin_prefetch(xrow[r], 0, 0);
  }

  wait_async0();
  __syncthreads();

  v8f ai[2] = {{}, {}}, ao[2] = {{}, {}}, au[2] = {{}, {}};
  for (int k0 = 0; k0 < EQ; k0 += 32) {
    v16bf ax0 = fragA_bf16(xrow[0], k0, laneHi);
    v16bf ax1 = fragA_bf16(xrow[1], k0, laneHi);
    v16bf wi = fragB_lds(tile, 0, laneN, 512, k0, laneHi);
    ai[0] = wmma_bf16(ax0, wi, ai[0]);  ai[1] = wmma_bf16(ax1, wi, ai[1]);
    v16bf wo = fragB_lds(tile, 1, laneN, 512, k0, laneHi);
    ao[0] = wmma_bf16(ax0, wo, ao[0]);  ao[1] = wmma_bf16(ax1, wo, ao[1]);
    v16bf wu = fragB_lds(tile, 2, laneN, 512, k0, laneHi);
    au[0] = wmma_bf16(ax0, wu, au[0]);  au[1] = wmma_bf16(ax1, wu, au[1]);
  }

  const float vbi = bi[nG], vbo = bo[nG], vbu = bu[nG];
#pragma unroll
  for (int r = 0; r < 2; ++r) {
#pragma unroll
    for (int v = 0; v < 8; ++v) {
      const int mG = mTile * 32 + r * 16 + v + laneHi * 8;
      const float iv = fsig(ai[r][v] + vbi);
      const float ov = fsig(ao[r][v] + vbo);
      const float uv = tanhf(au[r][v] + vbu);
      const float c  = iv * uv;
      const float h  = ov * tanhf(c);
      cOut[(size_t)mG * HQ + nG] = c;
      hOut[(size_t)mG * HQ + nG] = (bf16_t)h;
    }
  }
}

// ---------------------------------------------------------------------------
// Internal level. B tile = 8 gates x 16 cols; staged in two 64 KB K-halves.
// ---------------------------------------------------------------------------
__global__ void __launch_bounds__(128) internal_kernel(
    const int* __restrict__ word_ids, const bf16_t* __restrict__ embB,
    const bf16_t* __restrict__ Wall,
    const float* __restrict__ bi, const float* __restrict__ bfv,
    const float* __restrict__ bo, const float* __restrict__ bu,
    const bf16_t* __restrict__ hPrev, const float* __restrict__ cPrev,
    bf16_t* __restrict__ hOut, float* __restrict__ cOut,
    int lg)
{
  extern __shared__ char smem[];
  bf16_t* tile = (bf16_t*)smem;                 // [8][16][256] = 64 KB

  const int lane   = threadIdx.x & 31;
  const int wid    = threadIdx.x >> 5;
  const int laneN  = lane & 15;
  const int laneHi = lane >> 4;
  const int mTile  = blockIdx.x * 4 + wid;      // 32 rows per wave
  const int nTile  = blockIdx.y;
  const int nNodes = 1 << lg;
  const int start  = nNodes - 1;
  const int nG     = nTile * 16 + laneN;

  const bf16_t* xrow[2]; const bf16_t* h0row[2]; const bf16_t* h1row[2];
#pragma unroll
  for (int r = 0; r < 2; ++r) {
    const int rowA  = mTile * 32 + r * 16 + laneN;
    const int bA    = rowA >> lg;
    const int nodeA = rowA & (nNodes - 1);
    xrow[r]  = embB + (size_t)word_ids[bA * NSEQ + start + nodeA] * EQ;
    h0row[r] = hPrev + (size_t)((bA << (lg + 1)) + 2 * nodeA) * HQ;
    h1row[r] = h0row[r] + HQ;
    __builtin_prefetch(xrow[r],  0, 0);
    __builtin_prefetch(h0row[r], 0, 0);
  }

  v8f ai[2]  = {{}, {}}, ao[2]  = {{}, {}}, au[2]  = {{}, {}};
  v8f afx[2] = {{}, {}}, af0[2] = {{}, {}}, af1[2] = {{}, {}};

  for (int half = 0; half < 2; ++half) {
    const int kBase = half * 256;
    // Async-stage this K-half of the B tile: 8 gates x 16 rows x 256 K.
    for (int idx = threadIdx.x; idx < 8 * 16 * 32; idx += 128) {
      const int g   = idx >> 9;
      const int rem = idx & 511;
      const int row = rem >> 5;
      const int ck  = rem & 31;
      async_copy_b128(tile + ((size_t)(g * 16 + row) * 256 + ck * 8),
                      Wall + (size_t)g * MATQ + (size_t)(nTile * 16 + row) * 512
                           + kBase + ck * 8);
    }
    wait_async0();
    __syncthreads();

    for (int k0 = 0; k0 < 256; k0 += 32) {
      const int kg = kBase + k0;   // K offset into A rows
      v16bf ax0 = fragA_bf16(xrow[0],  kg, laneHi);
      v16bf ax1 = fragA_bf16(xrow[1],  kg, laneHi);
      v16bf a00 = fragA_bf16(h0row[0], kg, laneHi);
      v16bf a01 = fragA_bf16(h0row[1], kg, laneHi);
      v16bf a10 = fragA_bf16(h1row[0], kg, laneHi);
      v16bf a11 = fragA_bf16(h1row[1], kg, laneHi);

      v16bf wi = fragB_lds(tile, 0, laneN, 256, k0, laneHi);
      v16bf ui = fragB_lds(tile, 4, laneN, 256, k0, laneHi);
      ai[0] = wmma_bf16(ax0, wi, ai[0]);  ai[1] = wmma_bf16(ax1, wi, ai[1]);
      ai[0] = wmma_bf16(a00, ui, ai[0]);  ai[1] = wmma_bf16(a01, ui, ai[1]);
      ai[0] = wmma_bf16(a10, ui, ai[0]);  ai[1] = wmma_bf16(a11, ui, ai[1]);

      v16bf wo = fragB_lds(tile, 1, laneN, 256, k0, laneHi);
      v16bf uo = fragB_lds(tile, 5, laneN, 256, k0, laneHi);
      ao[0] = wmma_bf16(ax0, wo, ao[0]);  ao[1] = wmma_bf16(ax1, wo, ao[1]);
      ao[0] = wmma_bf16(a00, uo, ao[0]);  ao[1] = wmma_bf16(a01, uo, ao[1]);
      ao[0] = wmma_bf16(a10, uo, ao[0]);  ao[1] = wmma_bf16(a11, uo, ao[1]);

      v16bf wu = fragB_lds(tile, 2, laneN, 256, k0, laneHi);
      v16bf uu = fragB_lds(tile, 6, laneN, 256, k0, laneHi);
      au[0] = wmma_bf16(ax0, wu, au[0]);  au[1] = wmma_bf16(ax1, wu, au[1]);
      au[0] = wmma_bf16(a00, uu, au[0]);  au[1] = wmma_bf16(a01, uu, au[1]);
      au[0] = wmma_bf16(a10, uu, au[0]);  au[1] = wmma_bf16(a11, uu, au[1]);

      v16bf wf = fragB_lds(tile, 3, laneN, 256, k0, laneHi);
      v16bf uf = fragB_lds(tile, 7, laneN, 256, k0, laneHi);
      afx[0] = wmma_bf16(ax0, wf, afx[0]);  afx[1] = wmma_bf16(ax1, wf, afx[1]);
      af0[0] = wmma_bf16(a00, uf, af0[0]);  af0[1] = wmma_bf16(a01, uf, af0[1]);
      af1[0] = wmma_bf16(a10, uf, af1[0]);  af1[1] = wmma_bf16(a11, uf, af1[1]);
    }
    __syncthreads();   // tile about to be overwritten (or kernel ends)
  }

  const float vbi = bi[nG], vbf = bfv[nG], vbo = bo[nG], vbu = bu[nG];
#pragma unroll
  for (int r = 0; r < 2; ++r) {
#pragma unroll
    for (int v = 0; v < 8; ++v) {
      const int mG   = mTile * 32 + r * 16 + v + laneHi * 8;
      const int b    = mG >> lg;
      const int node = mG & (nNodes - 1);
      const int r0   = (b << (lg + 1)) + 2 * node;
      const float c0 = cPrev[(size_t)r0 * HQ + nG];
      const float c1 = cPrev[(size_t)(r0 + 1) * HQ + nG];

      const float iv = fsig(ai[r][v] + vbi);
      const float ov = fsig(ao[r][v] + vbo);
      const float uv = tanhf(au[r][v] + vbu);
      const float f0 = fsig(afx[r][v] + af0[r][v] + vbf);
      const float f1 = fsig(afx[r][v] + af1[r][v] + vbf);
      const float c  = iv * uv + f0 * c0 + f1 * c1;
      const float h  = ov * tanhf(c);
      cOut[(size_t)mG * HQ + nG] = c;
      hOut[(size_t)mG * HQ + nG] = (bf16_t)h;
    }
  }
}

// ---------------------------------------------------------------------------
// Output projection: out = h_root @ Wp + bp   (M=128, N=512, K=512)
// ---------------------------------------------------------------------------
__global__ void __launch_bounds__(128) proj_kernel(
    const bf16_t* __restrict__ hTop, const bf16_t* __restrict__ WpT,
    const float* __restrict__ bp, float* __restrict__ out)
{
  extern __shared__ char smem[];
  bf16_t* tile = (bf16_t*)smem;                 // [16][512] = 16 KB

  const int lane   = threadIdx.x & 31;
  const int wid    = threadIdx.x >> 5;
  const int laneN  = lane & 15;
  const int laneHi = lane >> 4;
  const int mTile  = blockIdx.x * 4 + wid;
  const int nTile  = blockIdx.y;
  const int nG     = nTile * 16 + laneN;

  for (int idx = threadIdx.x; idx < 16 * 64; idx += 128) {
    const int row = idx >> 6;
    const int ck  = idx & 63;
    async_copy_b128(tile + ((size_t)row * 512 + ck * 8),
                    WpT + (size_t)(nTile * 16 + row) * 512 + ck * 8);
  }

  const bf16_t* arow[2] = { hTop + (size_t)(mTile * 32 + laneN) * HQ,
                            hTop + (size_t)(mTile * 32 + 16 + laneN) * HQ };
  wait_async0();
  __syncthreads();

  v8f acc[2] = {{}, {}};
  for (int k0 = 0; k0 < HQ; k0 += 32) {
    v16bf bfrag = fragB_lds(tile, 0, laneN, 512, k0, laneHi);
    acc[0] = wmma_bf16(fragA_bf16(arow[0], k0, laneHi), bfrag, acc[0]);
    acc[1] = wmma_bf16(fragA_bf16(arow[1], k0, laneHi), bfrag, acc[1]);
  }
  const float vbp = bp[nG];
#pragma unroll
  for (int r = 0; r < 2; ++r) {
#pragma unroll
    for (int v = 0; v < 8; ++v) {
      const int mG = mTile * 32 + r * 16 + v + laneHi * 8;
      out[(size_t)mG * EQ + nG] = acc[r][v] + vbp;
    }
  }
}

// ---------------------------------------------------------------------------
extern "C" void kernel_launch(void* const* d_in, const int* in_sizes, int n_in,
                              void* d_out, int out_size, void* d_ws, size_t ws_size,
                              hipStream_t stream) {
  (void)in_sizes; (void)n_in; (void)out_size; (void)ws_size;
  const int*   word_ids = (const int*)  d_in[0];
  const float* emb      = (const float*)d_in[1];
  const float* Wi = (const float*)d_in[2];
  const float* bi = (const float*)d_in[3];
  const float* Ui = (const float*)d_in[4];
  const float* Wf = (const float*)d_in[5];
  const float* bf = (const float*)d_in[6];
  const float* Uf = (const float*)d_in[7];
  const float* Wo = (const float*)d_in[8];
  const float* bo = (const float*)d_in[9];
  const float* Uo = (const float*)d_in[10];
  const float* Wu = (const float*)d_in[11];
  const float* bu = (const float*)d_in[12];
  const float* Uu = (const float*)d_in[13];
  const float* Wp = (const float*)d_in[14];
  const float* bp = (const float*)d_in[15];

  // Workspace carve-up (~188 MB total).
  char* ws = (char*)d_ws;
  size_t off = 0;
  bf16_t* Wall = (bf16_t*)(ws + off); off += (size_t)8 * MATQ * sizeof(bf16_t); // WxT|UT
  bf16_t* WpT  = (bf16_t*)(ws + off); off += (size_t)MATQ * sizeof(bf16_t);
  bf16_t* embB = (bf16_t*)(ws + off); off += (size_t)32000 * EQ * sizeof(bf16_t);
  bf16_t* hA   = (bf16_t*)(ws + off); off += (size_t)BQ * 256 * HQ * sizeof(bf16_t);
  bf16_t* hB   = (bf16_t*)(ws + off); off += (size_t)BQ * 128 * HQ * sizeof(bf16_t);
  float*  cA   = (float*) (ws + off); off += (size_t)BQ * 256 * HQ * sizeof(float);
  float*  cB   = (float*) (ws + off); off += (size_t)BQ * 128 * HQ * sizeof(float);

  // 1) Pack weights + embedding table to bf16.
  cvt_weights_kernel<<<dim3(9 * MATQ / 256), 256, 0, stream>>>(
      Wi, Wo, Wu, Wf, Ui, Uo, Uu, Uf, Wp, Wall, WpT);
  cvt_emb_kernel<<<dim3(32000 * EQ / 8 / 256), 256, 0, stream>>>(emb, embB);

  // 2) Leaf level (lvl 8): 32768 rows -> even-parity buffers (A).
  leaf_kernel<<<dim3(256, 32), 128, 3 * 16 * 512 * sizeof(bf16_t), stream>>>(
      word_ids, embB, Wall, bi, bo, bu, hA, cA);

  // 3) Internal levels 7..0, ping-ponging by level parity.
  for (int lvl = 7; lvl >= 0; --lvl) {
    bf16_t*       hDst = (lvl & 1) ? hB : hA;
    float*        cDst = (lvl & 1) ? cB : cA;
    const bf16_t* hSrc = (lvl & 1) ? hA : hB;
    const float*  cSrc = (lvl & 1) ? cA : cB;
    const int gridX = 1 << lvl;          // 128 rows per 4-wave block
    internal_kernel<<<dim3(gridX, 32), 128, 8 * 16 * 256 * sizeof(bf16_t), stream>>>(
        word_ids, embB, Wall, bi, bf, bo, bu, hSrc, cSrc, hDst, cDst, lvl);
  }

  // 4) Projection from root h (level 0 lives in hA).
  proj_kernel<<<dim3(1, 32), 128, 16 * 512 * sizeof(bf16_t), stream>>>(
      hA, WpT, bp, (float*)d_out);
}
